// TftDeepMomentumNetworkModel_18030272708889
// MI455X (gfx1250) — matrix operand order repaired and checked by reading, not compiled, not run on hardware
//
#include <hip/hip_runtime.h>
#include <hip/hip_bf16.h>
#include <math.h>

typedef __attribute__((ext_vector_type(16))) __bf16 v16bf;
typedef __attribute__((ext_vector_type(8)))  float  v8f;
typedef __attribute__((ext_vector_type(4)))  int    vi4;

#define H_DIM 256
#define T_DIM 512
#define B_DIM 64
#define DK_DIM 64

// ---- CDNA5 async global->LDS path (guarded; falls back to sync loads) ----
#if defined(__HIP_DEVICE_COMPILE__) && __has_builtin(__builtin_amdgcn_global_load_async_to_lds_b128)
#define ASYNC_LDS 1
#else
#define ASYNC_LDS 0
#endif

// Builtin signature (probe-confirmed): (v4i AS1*, v4i AS3*, imm offset, imm cpol)
#define GLOBAL_AS4(p) ((__attribute__((address_space(1))) vi4*)(p))
#define LDS_AS4(p)    ((__attribute__((address_space(3))) vi4*)(p))

__device__ __forceinline__ void wait_async_all()
{
#if defined(__HIP_DEVICE_COMPILE__)
#if __has_builtin(__builtin_amdgcn_s_wait_asynccnt)
    __builtin_amdgcn_s_wait_asynccnt(0);
#else
    asm volatile("s_wait_asynccnt 0" ::: "memory");
#endif
#endif
}

__device__ __forceinline__ float sigf(float x) { return 1.f / (1.f + __expf(-x)); }

// ---------------------------------------------------------------------------
// Weight prep: f32 -> bf16 (optionally transposed so GEMM-B is always K x N)
// ---------------------------------------------------------------------------
__global__ void k_cvt(const float* __restrict__ src, __bf16* __restrict__ dst, int n)
{
    int i = blockIdx.x * 256 + threadIdx.x;
    if (i < n) dst[i] = (__bf16)src[i];
}

// src is (rows x cols) row-major; dst[k*rows + n] = src[n*cols + k]  (K=cols, N=rows)
__global__ void k_cvt_t(const float* __restrict__ src, __bf16* __restrict__ dst,
                        int rows, int cols)
{
    int i = blockIdx.x * 256 + threadIdx.x;
    if (i < rows * cols) {
        int n = i / cols, k = i % cols;
        dst[(size_t)k * rows + n] = (__bf16)src[i];
    }
}

// ---------------------------------------------------------------------------
// Featurize: historical = (sum_{r=1..7} x_reg*reg_w + reg_b + emb0 + emb1)/9
//            static_ctx = tr[:,0,0,:]
// ---------------------------------------------------------------------------
__global__ void k_featurize(const float* __restrict__ x_reg, const int* __restrict__ x_cat,
                            const float* __restrict__ reg_w, const float* __restrict__ reg_b,
                            const float* __restrict__ emb0, const float* __restrict__ emb1,
                            float* __restrict__ historical, float* __restrict__ static_ctx)
{
    int bt = blockIdx.x; int h = threadIdx.x;
    int b = bt >> 9, t = bt & (T_DIM - 1);
    const float* xr = x_reg + (size_t)bt * 8;
    float s = 0.f;
#pragma unroll
    for (int r = 1; r < 8; ++r) s += xr[r] * reg_w[r * H_DIM + h] + reg_b[r * H_DIM + h];
    int c0 = x_cat[(size_t)bt * 2 + 0], c1 = x_cat[(size_t)bt * 2 + 1];
    s += emb0[c0 * H_DIM + h] + emb1[c1 * H_DIM + h];
    historical[(size_t)bt * H_DIM + h] = s * (1.f / 9.f);
    if (t == 0) static_ctx[b * H_DIM + h] = xr[0] * reg_w[h] + reg_b[h];
}

// ---------------------------------------------------------------------------
// bf16 WMMA GEMM: D = epi(A(MxK,f32) * Bw(KxN,bf16) [+bias][+ctx][+addrow])
// Block tile 64x32, 8 waves x one 16x16 C tile, v_wmma_f32_16x16x32_bf16.
// Tiles staged in LDS via global_load_async_to_lds_b128 when available.
// ACT: 0 none, 1 ELU, 2 sigmoid. Output: f32 D, or bf16 D16 (d16T: store N-major).
// ---------------------------------------------------------------------------
template<int ACT>
__global__ void k_gemm(const float* __restrict__ A, const __bf16* __restrict__ Bw,
                       const float* __restrict__ bias,
                       const float* __restrict__ ctx, int ctxShift,
                       const float* __restrict__ addrow, long addStride,
                       float* __restrict__ D, __bf16* __restrict__ D16, int d16T,
                       int M, int N, int K)
{
    __shared__ float  Asf[64][32];
    __shared__ __bf16 Bs[32][32];
    const int tid  = threadIdx.x;
    const int lane = tid & 31, wave = tid >> 5;
    const int l16  = lane & 15, hs = lane >> 4;
    const int mt   = wave & 3,  nt = wave >> 2;
    const int blockN = blockIdx.x * 32;
    const int blockM = blockIdx.y * 64;

    v8f acc = {};
    for (int kk = 0; kk < K; kk += 32) {
#if ASYNC_LDS
        // A tile: 64x32 f32 = 512 x 16B segments
        for (int i = tid; i < 512; i += 256) {
            int r = i >> 3, c0 = (i & 7) * 4;
            __builtin_amdgcn_global_load_async_to_lds_b128(
                GLOBAL_AS4(&A[(size_t)(blockM + r) * K + kk + c0]),
                LDS_AS4(&Asf[r][c0]), 0, 0);
        }
        // B tile: 32x32 bf16 = 128 x 16B segments
        if (tid < 128) {
            int r = tid >> 2, c0 = (tid & 3) * 8;
            __builtin_amdgcn_global_load_async_to_lds_b128(
                GLOBAL_AS4(&Bw[(size_t)(kk + r) * N + blockN + c0]),
                LDS_AS4(&Bs[r][c0]), 0, 0);
        }
        wait_async_all();
#else
        for (int i = tid; i < 64 * 32; i += 256) {
            int r = i >> 5, c = i & 31;
            Asf[r][c] = A[(size_t)(blockM + r) * K + kk + c];
        }
        for (int i = tid; i < 32 * 32; i += 256) {
            int r = i >> 5, c = i & 31;
            Bs[r][c] = Bw[(size_t)(kk + r) * N + blockN + c];
        }
        if (kk + 32 < K)
            __builtin_prefetch(&A[(size_t)(blockM + (tid >> 2)) * K + kk + 32], 0, 1);
#endif
        __syncthreads();

        // Fragments per CDNA5 ISA 7.12.2 layouts (wave32).
        v16bf a, bq;
        const int am = mt * 16 + l16;
#pragma unroll
        for (int j = 0; j < 4; ++j) {
            int k0 = hs * 8 + 2 * j;          // lanes0-15: K0..7 ; lanes16-31: K8..15
            a[2 * j]     = (__bf16)Asf[am][k0];
            a[2 * j + 1] = (__bf16)Asf[am][k0 + 1];
            int k1 = 16 + hs * 8 + 2 * j;     // K16..23 / K24..31
            a[2 * j + 8] = (__bf16)Asf[am][k1];
            a[2 * j + 9] = (__bf16)Asf[am][k1 + 1];
        }
        const int bn = nt * 16 + l16;
#pragma unroll
        for (int j = 0; j < 8; ++j) {
            int kq = (j >> 2) * 16 + hs * 8 + (j & 3) * 2;  // lanes0-15: K0..7,16..23 ; lanes16-31: K8..15,24..31
            bq[2 * j]     = Bs[kq][bn];
            bq[2 * j + 1] = Bs[kq + 1][bn];
        }
        acc = __builtin_amdgcn_wmma_f32_16x16x32_bf16(false, a, false, bq, (short)0, acc,
                                                      false, false);
        __syncthreads();
    }

    const int col = blockN + nt * 16 + l16;
#pragma unroll
    for (int j = 0; j < 8; ++j) {
        int row = blockM + mt * 16 + hs * 8 + j;   // C layout: VGPR j -> M=j (+8 hi half)
        float v = acc[j];
        if (bias)   v += bias[col];
        if (ctx)    v += ctx[(size_t)(row >> ctxShift) * H_DIM + col];
        if (addrow) v += addrow[(size_t)row * addStride + col];
        if (ACT == 1) v = v > 0.f ? v : (__expf(v) - 1.f);
        if (ACT == 2) v = sigf(v);
        if (D16) {
            if (d16T) D16[(size_t)col * M + row] = (__bf16)v;
            else      D16[(size_t)row * N + col] = (__bf16)v;
        } else {
            D[(size_t)row * N + col] = v;
        }
    }
}

// ---------------------------------------------------------------------------
// GRN tail: y = x + h2*g; layernorm(y)*gamma + beta [+ post]
// ---------------------------------------------------------------------------
__global__ void k_grn_finish(const float* __restrict__ x, const float* __restrict__ h2,
                             const float* __restrict__ g, const float* __restrict__ gamma,
                             const float* __restrict__ beta, const float* __restrict__ post,
                             float* __restrict__ out)
{
    __shared__ float red[256];
    int row = blockIdx.x, tid = threadIdx.x;
    size_t idx = (size_t)row * H_DIM + tid;
    float y = x[idx] + h2[idx] * g[idx];
    red[tid] = y; __syncthreads();
    for (int s = 128; s > 0; s >>= 1) { if (tid < s) red[tid] += red[tid + s]; __syncthreads(); }
    float m = red[0] * (1.f / 256.f); __syncthreads();
    float d = y - m;
    red[tid] = d * d; __syncthreads();
    for (int s = 128; s > 0; s >>= 1) { if (tid < s) red[tid] += red[tid + s]; __syncthreads(); }
    float var = red[0] * (1.f / 256.f);
    float r = d * rsqrtf(var + 1e-5f) * gamma[tid] + beta[tid];
    if (post) r += post[idx];
    out[idx] = r;
}

// ---------------------------------------------------------------------------
// LSTM pointwise gates (gbuf 64x1024, order i,f,g,o)
// ---------------------------------------------------------------------------
__global__ void k_lstm_gate(const float* __restrict__ gbuf, float* __restrict__ hstate,
                            float* __restrict__ cstate, float* __restrict__ lstm_out, int t)
{
    int b = blockIdx.x, h = threadIdx.x;
    const float* gr = gbuf + (size_t)b * 1024;
    float gi = gr[h], gf = gr[256 + h], gg = gr[512 + h], go = gr[768 + h];
    float c = cstate[b * H_DIM + h];
    c = sigf(gf) * c + sigf(gi) * tanhf(gg);
    float hh = sigf(go) * tanhf(c);
    cstate[b * H_DIM + h] = c;
    hstate[b * H_DIM + h] = hh;
    lstm_out[((size_t)b * T_DIM + t) * H_DIM + h] = hh;
}

__global__ void k_biassum(const float* __restrict__ a, const float* __restrict__ b,
                          float* __restrict__ o)
{
    int i = blockIdx.x * 256 + threadIdx.x;
    o[i] = a[i] + b[i];
}

// ---------------------------------------------------------------------------
// Attention: block per (head, batch, 16-query tile). q/k bf16 row-major,
// v bf16 stored transposed: vT[n][b*T + t]. WMMA QK^T -> LDS, causal softmax,
// WMMA A*V -> headbuf (f32).
// ---------------------------------------------------------------------------
__global__ void k_attention(const __bf16* __restrict__ qb, const __bf16* __restrict__ kb,
                            const __bf16* __restrict__ vT, float* __restrict__ headbuf)
{
    const int qt = blockIdx.x, b = blockIdx.y, h = blockIdx.z;
    __shared__ float  sc[16][T_DIM];
    __shared__ __bf16 qs[16][DK_DIM];
    __shared__ float  red[16][17];
    const int tid  = threadIdx.x;
    const int lane = tid & 31, wave = tid >> 5;
    const int l16  = lane & 15, hs = lane >> 4;
    const int Lk   = (qt + 1) * 16;
    const int Lk32 = (Lk + 31) & ~31;

    for (int i = tid; i < 16 * T_DIM; i += 256) (&sc[0][0])[i] = 0.f;
    const size_t qkbase = (size_t)(h * B_DIM + b) * T_DIM;
#if ASYNC_LDS
    if (tid < 128) {    // 16x64 bf16 = 128 x 16B segments
        int r = tid >> 3, c0 = (tid & 7) * 8;
        __builtin_amdgcn_global_load_async_to_lds_b128(
            GLOBAL_AS4(&qb[(qkbase + qt * 16 + r) * DK_DIM + c0]),
            LDS_AS4(&qs[r][c0]), 0, 0);
    }
    wait_async_all();
#else
    for (int i = tid; i < 16 * DK_DIM; i += 256) {
        int r = i >> 6, c = i & 63;
        qs[r][c] = qb[(qkbase + qt * 16 + r) * DK_DIM + c];
    }
#endif
    __syncthreads();

    // ---- scores: S = Q K^T / 8, causal mask ----
    for (int kt = wave; kt <= qt; kt += 8) {
        v8f acc = {};
#pragma unroll
        for (int kk = 0; kk < 64; kk += 32) {
            v16bf a, bq;
#pragma unroll
            for (int j = 0; j < 4; ++j) {
                int k0 = kk + hs * 8 + 2 * j;
                a[2 * j]     = qs[l16][k0];
                a[2 * j + 1] = qs[l16][k0 + 1];
                int k1 = kk + 16 + hs * 8 + 2 * j;
                a[2 * j + 8] = qs[l16][k1];
                a[2 * j + 9] = qs[l16][k1 + 1];
            }
            const size_t krow = (qkbase + kt * 16 + l16) * DK_DIM;
#pragma unroll
            for (int j = 0; j < 8; ++j) {
                int kq = kk + (j >> 2) * 16 + hs * 8 + (j & 3) * 2;
                bq[2 * j]     = kb[krow + kq];
                bq[2 * j + 1] = kb[krow + kq + 1];
            }
            acc = __builtin_amdgcn_wmma_f32_16x16x32_bf16(false, a, false, bq, (short)0, acc,
                                                          false, false);
        }
#pragma unroll
        for (int j = 0; j < 8; ++j) {
            int m = hs * 8 + j;
            int colk = kt * 16 + l16;
            float s = acc[j] * 0.125f;                 // 1/sqrt(DK)
            if (colk > qt * 16 + m) s = -1.0e9f;       // causal
            sc[m][colk] = s;
        }
    }
    __syncthreads();

    // ---- softmax over Lk entries per row (16 threads/row) ----
    {
        const int r = tid >> 4, c0 = tid & 15;
        float mx = -3.0e38f;
        for (int c = c0; c < Lk; c += 16) mx = fmaxf(mx, sc[r][c]);
        red[r][c0] = mx; __syncthreads();
        if (c0 == 0) {
            float m2 = red[r][0];
            for (int i = 1; i < 16; ++i) m2 = fmaxf(m2, red[r][i]);
            red[r][16] = m2;
        }
        __syncthreads();
        mx = red[r][16];
        float sum = 0.f;
        for (int c = c0; c < Lk; c += 16) { float e = __expf(sc[r][c] - mx); sc[r][c] = e; sum += e; }
        red[r][c0] = sum; __syncthreads();
        if (c0 == 0) {
            float s2 = 0.f;
            for (int i = 0; i < 16; ++i) s2 += red[r][i];
            red[r][16] = 1.f / s2;
        }
        __syncthreads();
        float inv = red[r][16];
        for (int c = c0; c < Lk; c += 16) sc[r][c] *= inv;
    }
    __syncthreads();

    // ---- head tile = attn(16 x Lk32) * V(Lk32 x 64); waves 0..3 each 16 cols ----
    if (wave < 4) {
        const int nt = wave;
        const int n = nt * 16 + l16;
        const size_t vrow = (size_t)n * (B_DIM * T_DIM) + (size_t)b * T_DIM;
        v8f acc = {};
        for (int kk = 0; kk < Lk32; kk += 32) {
            v16bf a, bq;
#pragma unroll
            for (int j = 0; j < 4; ++j) {
                int k0 = kk + hs * 8 + 2 * j;
                a[2 * j]     = (__bf16)sc[l16][k0];
                a[2 * j + 1] = (__bf16)sc[l16][k0 + 1];
                int k1 = kk + 16 + hs * 8 + 2 * j;
                a[2 * j + 8] = (__bf16)sc[l16][k1];
                a[2 * j + 9] = (__bf16)sc[l16][k1 + 1];
            }
#pragma unroll
            for (int j = 0; j < 8; ++j) {
                int kpos = kk + (j >> 2) * 16 + hs * 8 + (j & 3) * 2;
                bq[2 * j]     = vT[vrow + kpos];
                bq[2 * j + 1] = vT[vrow + kpos + 1];
            }
            acc = __builtin_amdgcn_wmma_f32_16x16x32_bf16(false, a, false, bq, (short)0, acc,
                                                          false, false);
        }
#pragma unroll
        for (int j = 0; j < 8; ++j) {
            int m = hs * 8 + j;
            headbuf[(qkbase + qt * 16 + m) * DK_DIM + nt * 16 + l16] = acc[j];
        }
    }
}

__global__ void k_head_mean(const float* __restrict__ headbuf, float* __restrict__ out)
{
    size_t i = (size_t)blockIdx.x * 256 + threadIdx.x;       // over B*T*DK
    const size_t HS = (size_t)B_DIM * T_DIM * DK_DIM;
    out[i] = 0.25f * (headbuf[i] + headbuf[i + HS] + headbuf[i + 2 * HS] + headbuf[i + 3 * HS]);
}

// out = tanh(x . out_w + out_b), one row per block
__global__ void k_out(const float* __restrict__ x, const float* __restrict__ w,
                      const float* __restrict__ bscalar, float* __restrict__ out)
{
    __shared__ float red[256];
    int row = blockIdx.x, tid = threadIdx.x;
    red[tid] = x[(size_t)row * H_DIM + tid] * w[tid];
    __syncthreads();
    for (int s = 128; s > 0; s >>= 1) { if (tid < s) red[tid] += red[tid + s]; __syncthreads(); }
    if (tid == 0) out[row] = tanhf(red[0] + bscalar[0]);
}

// ---------------------------------------------------------------------------
// Host orchestration
// ---------------------------------------------------------------------------
struct GrnP {
    const float *b1, *b2, *bg, *gamma, *beta;   // f32 vectors
    const __bf16 *w1, *w2, *wg;                 // pre-converted bf16 weights
};

static void run_grn(const float* X, int M, const GrnP& p, const float* ctx, int ctxShift,
                    const float* post, float* out, float* t1, float* t2, float* t3,
                    hipStream_t s)
{
    dim3 g(H_DIM / 32, M / 64);
    k_gemm<1><<<g, 256, 0, s>>>(X, p.w1, p.b1, ctx, ctxShift, nullptr, 0,
                                t1, nullptr, 0, M, H_DIM, H_DIM);
    k_gemm<0><<<g, 256, 0, s>>>(t1, p.w2, p.b2, nullptr, 0, nullptr, 0,
                                t2, nullptr, 0, M, H_DIM, H_DIM);
    k_gemm<2><<<g, 256, 0, s>>>(t1, p.wg, p.bg, nullptr, 0, nullptr, 0,
                                t3, nullptr, 0, M, H_DIM, H_DIM);
    k_grn_finish<<<M, 256, 0, s>>>(X, t2, t3, p.gamma, p.beta, post, out);
}

extern "C" void kernel_launch(void* const* d_in, const int* in_sizes, int n_in,
                              void* d_out, int out_size, void* d_ws, size_t ws_size,
                              hipStream_t stream)
{
    const size_t BT  = (size_t)B_DIM * T_DIM;          // 32768
    const size_t BIG = BT * H_DIM;                     // 8,388,608 floats

    // ---- workspace partition (float units; bf16 regions carved as 2/float) ----
    float* W = (float*)d_ws;
    size_t o = 0;
    auto alloc  = [&](size_t n) { float* p = W + o; o += (n + 3) & ~(size_t)3; return p; };
    auto allocb = [&](size_t nbf16) { return (__bf16*)alloc(nbf16 / 2); };

    float* historical = alloc(BIG);       // later reused as attn_out
    float* lstm_out   = alloc(BIG);       // later reused as transformer_out
    float* temporal   = alloc(BIG);
    float* enriched   = alloc(BIG);
    float* t1 = alloc(BIG);
    float* t2 = alloc(BIG);
    float* t3 = alloc(BIG);
    float* xg = alloc(BT * 1024);         // 33.5M floats; region reused post-LSTM:
    __bf16* qb16 = (__bf16*)xg;                               // 4*BT*64 bf16
    __bf16* kb16 = (__bf16*)(xg + 4194304);                   // 4*BT*64 bf16
    __bf16* vT16 = (__bf16*)(xg + 8388608);                   // BT*64 bf16 (transposed)
    float* headbuf  = xg + 9437184;                           // 4*BT*64 f32
    float* headmean = xg + 17825792;                          // BT*64 f32
    float* static_ctx = alloc(B_DIM * H_DIM);
    float* sc_enr     = alloc(B_DIM * H_DIM);
    float* hstate     = alloc(B_DIM * H_DIM);
    float* cstate     = alloc(B_DIM * H_DIM);
    float* ctxproj    = alloc(B_DIM * H_DIM);
    float* s1 = alloc(B_DIM * H_DIM);
    float* s2 = alloc(B_DIM * H_DIM);
    float* s3 = alloc(B_DIM * H_DIM);
    float* bsum  = alloc(1024);
    float* gstep = alloc(B_DIM * 1024);
    // bf16 weight area
    __bf16* grnw[6][3];
    for (int g = 0; g < 6; ++g)
        for (int m = 0; m < 3; ++m) grnw[g][m] = allocb(H_DIM * H_DIM);
    __bf16* wc16   = allocb(H_DIM * H_DIM);
    __bf16* WihT16 = allocb(H_DIM * 4 * H_DIM);
    __bf16* WhhT16 = allocb(H_DIM * 4 * H_DIM);
    __bf16* wq16   = allocb(4 * H_DIM * DK_DIM);
    __bf16* wk16   = allocb(4 * H_DIM * DK_DIM);
    __bf16* wv16   = allocb(H_DIM * DK_DIM);
    __bf16* wo16   = allocb(DK_DIM * H_DIM);
    if (ws_size < o * sizeof(float)) return;           // workspace too small

    // ---- inputs (setup_inputs dict order, nested dicts flattened in order) ----
    const float* x_reg = (const float*)d_in[0];
    const int*   x_cat = (const int*)d_in[1];
    const float* reg_w = (const float*)d_in[2];
    const float* reg_b = (const float*)d_in[3];
    const float* emb0  = (const float*)d_in[4];
    const float* emb1  = (const float*)d_in[5];
    // grn_vs at 6..13 is dead code in the reference -> skipped
    const int grn_base[6] = {14, 22, 30, 38, 46, 55}; // enr, h, c, temporal, se, dec
    const float* wc = (const float*)d_in[54];
    const float* Wih  = (const float*)d_in[63];
    const float* Whh  = (const float*)d_in[64];
    const float* bih  = (const float*)d_in[65];
    const float* bhh  = (const float*)d_in[66];
    const float* wq   = (const float*)d_in[67];
    const float* wk   = (const float*)d_in[68];
    const float* wv   = (const float*)d_in[69];
    const float* wo   = (const float*)d_in[70];
    const float* out_w = (const float*)d_in[71];
    const float* out_b = (const float*)d_in[72];

    GrnP grn[6];
    for (int g = 0; g < 6; ++g) {
        int base = grn_base[g];
        grn[g].b1 = (const float*)d_in[base + 1];
        grn[g].b2 = (const float*)d_in[base + 3];
        grn[g].bg = (const float*)d_in[base + 5];
        grn[g].gamma = (const float*)d_in[base + 6];
        grn[g].beta  = (const float*)d_in[base + 7];
        grn[g].w1 = grnw[g][0]; grn[g].w2 = grnw[g][1]; grn[g].wg = grnw[g][2];
    }

    // ---- 0) weight conversion to bf16 (Wih/Whh transposed to K x N) ----
    auto cvt = [&](const float* src, __bf16* dst, int n) {
        k_cvt<<<(n + 255) / 256, 256, 0, stream>>>(src, dst, n);
    };
    for (int g = 0; g < 6; ++g) {
        int base = grn_base[g];
        cvt((const float*)d_in[base + 0], grnw[g][0], H_DIM * H_DIM);
        cvt((const float*)d_in[base + 2], grnw[g][1], H_DIM * H_DIM);
        cvt((const float*)d_in[base + 4], grnw[g][2], H_DIM * H_DIM);
    }
    cvt(wc, wc16, H_DIM * H_DIM);
    k_cvt_t<<<(4 * H_DIM * H_DIM + 255) / 256, 256, 0, stream>>>(Wih, WihT16, 4 * H_DIM, H_DIM);
    k_cvt_t<<<(4 * H_DIM * H_DIM + 255) / 256, 256, 0, stream>>>(Whh, WhhT16, 4 * H_DIM, H_DIM);
    cvt(wq, wq16, 4 * H_DIM * DK_DIM);
    cvt(wk, wk16, 4 * H_DIM * DK_DIM);
    cvt(wv, wv16, H_DIM * DK_DIM);
    cvt(wo, wo16, DK_DIM * H_DIM);

    // ---- 1) featurize ----
    k_featurize<<<(int)BT, 256, 0, stream>>>(x_reg, x_cat, reg_w, reg_b, emb0, emb1,
                                             historical, static_ctx);

    // ---- 2) static GRNs ----
    run_grn(static_ctx, B_DIM, grn[0], nullptr, 0, nullptr, sc_enr, s1, s2, s3, stream);
    run_grn(static_ctx, B_DIM, grn[1], nullptr, 0, nullptr, hstate, s1, s2, s3, stream);
    run_grn(static_ctx, B_DIM, grn[2], nullptr, 0, nullptr, cstate, s1, s2, s3, stream);

    // ctxproj = sc_enr @ wc
    k_gemm<0><<<dim3(H_DIM / 32, 1), 256, 0, stream>>>(
        sc_enr, wc16, nullptr, nullptr, 0, nullptr, 0, ctxproj, nullptr, 0,
        B_DIM, H_DIM, H_DIM);

    // ---- 3) LSTM input projection: xg = historical @ Wih^T + (bih + bhh) ----
    k_biassum<<<4, 256, 0, stream>>>(bih, bhh, bsum);
    k_gemm<0><<<dim3(1024 / 32, (int)(BT / 64)), 256, 0, stream>>>(
        historical, WihT16, bsum, nullptr, 0, nullptr, 0, xg, nullptr, 0,
        (int)BT, 1024, H_DIM);

    // ---- 4) LSTM recurrence (sequential in T) ----
    for (int t = 0; t < T_DIM; ++t) {
        k_gemm<0><<<dim3(1024 / 32, 1), 256, 0, stream>>>(
            hstate, WhhT16, nullptr, nullptr, 0, xg + (size_t)t * 1024, (long)T_DIM * 1024,
            gstep, nullptr, 0, B_DIM, 1024, H_DIM);
        k_lstm_gate<<<B_DIM, 256, 0, stream>>>(gstep, hstate, cstate, lstm_out, t);
    }

    // ---- 5) temporal = grn_temporal(lstm_out) + historical ----
    run_grn(lstm_out, (int)BT, grn[3], nullptr, 0, historical, temporal, t1, t2, t3, stream);

    // ---- 6) enriched = grn_se(temporal, ctx broadcast over T) ----
    run_grn(temporal, (int)BT, grn[4], ctxproj, 9 /*row>>9 = b*/, nullptr, enriched,
            t1, t2, t3, stream);

    // ---- 7) q, k, v projections (bf16 outputs; v transposed) ----
    for (int h = 0; h < 4; ++h) {
        k_gemm<0><<<dim3(DK_DIM / 32, (int)(BT / 64)), 256, 0, stream>>>(
            enriched, wq16 + (size_t)h * H_DIM * DK_DIM, nullptr, nullptr, 0, nullptr, 0,
            nullptr, qb16 + (size_t)h * BT * DK_DIM, 0, (int)BT, DK_DIM, H_DIM);
        k_gemm<0><<<dim3(DK_DIM / 32, (int)(BT / 64)), 256, 0, stream>>>(
            enriched, wk16 + (size_t)h * H_DIM * DK_DIM, nullptr, nullptr, 0, nullptr, 0,
            nullptr, kb16 + (size_t)h * BT * DK_DIM, 0, (int)BT, DK_DIM, H_DIM);
    }
    k_gemm<0><<<dim3(DK_DIM / 32, (int)(BT / 64)), 256, 0, stream>>>(
        enriched, wv16, nullptr, nullptr, 0, nullptr, 0,
        nullptr, vT16, 1 /*transposed store*/, (int)BT, DK_DIM, H_DIM);

    // ---- 8) causal attention per (head, batch, q-tile) ----
    k_attention<<<dim3(T_DIM / 16, B_DIM, 4), 256, 0, stream>>>(qb16, kb16, vT16, headbuf);
    k_head_mean<<<(int)(BT * DK_DIM / 256), 256, 0, stream>>>(headbuf, headmean);

    // ---- 9) attn_out = headmean @ wo + enriched (reuse `historical`) ----
    float* attn_out = historical;
    k_gemm<0><<<dim3(H_DIM / 32, (int)(BT / 64)), 256, 0, stream>>>(
        headmean, wo16, nullptr, nullptr, 0, enriched, (long)H_DIM,
        attn_out, nullptr, 0, (int)BT, H_DIM, DK_DIM);

    // ---- 10) transformer_out = grn_dec(attn_out) + temporal (reuse lstm_out) ----
    float* trans_out = lstm_out;
    run_grn(attn_out, (int)BT, grn[5], nullptr, 0, temporal, trans_out, t1, t2, t3, stream);

    // ---- 11) out = tanh(trans_out @ out_w + out_b) ----
    k_out<<<(int)BT, 256, 0, stream>>>(trans_out, out_w, out_b, (float*)d_out);
}
